// RGBChannelAttention_24558622998808
// MI455X (gfx1250) — compile-verified, matrix operand order checked by reading, not run on hardware
//
#include <hip/hip_runtime.h>
#include <hip/hip_bf16.h>

typedef __attribute__((ext_vector_type(16))) _Float16 v16h;
typedef __attribute__((ext_vector_type(8)))  float    v8f;

#define BSZ 2
#define CCH 256
#define NTOK 4096            // 64*64
#define ATTN_SCALE 0.0625f   // 1/sqrt(256)
#define EPSV 1e-5f

// LDS layout for k_attn (bytes)
#define LDS_SH    0u            // S strip f16 [16][4096]       = 131072
#define LDS_QS    131072u       // Q stage f16 [16][256]        = 8192
#define LDS_KST   139264u       // 8 waves x 2 bufs x 8192      = 131072
#define LDS_ROWL  270336u       // 16 floats                    = 64
#define LDS_TOTAL 270400u

// ---------------------------------------------------------------------------
// Tensor Data Mover: 2D f16 tile DMA global -> LDS (D# per ISA 8.3/8.4).
// This toolchain exposes the 6-arg builtin:
//   (uint32x4 g0, int32x8 g1, int32x4 g2, int32x4 g3, int32x8 extra, i32 cpol)
// Guarded: falls back to direct global gathers when builtin is absent.
// ---------------------------------------------------------------------------
#if __has_builtin(__builtin_amdgcn_tensor_load_to_lds) && \
    __has_builtin(__builtin_amdgcn_s_wait_tensorcnt)
#define HAVE_TDM 1
typedef __attribute__((ext_vector_type(4))) unsigned int u32x4;
typedef __attribute__((ext_vector_type(8))) int i32x8;
typedef __attribute__((ext_vector_type(4))) int i32x4;

__device__ __forceinline__ void tdm_load_2d_f16(unsigned lds_byte_off, const void* gsrc,
                                                unsigned tile_w, unsigned tile_h,
                                                unsigned row_stride_elems) {
  unsigned long long ga = (unsigned long long)(uintptr_t)gsrc;
  // group0: count=1 | lds_addr | global_addr[56:0] | type=2
  unsigned g00 = 1u;
  unsigned g01 = (unsigned)__builtin_amdgcn_readfirstlane((int)lds_byte_off);
  unsigned g02 = (unsigned)__builtin_amdgcn_readfirstlane((int)(unsigned)ga);
  unsigned g03 = (unsigned)__builtin_amdgcn_readfirstlane(
                     (int)(((unsigned)(ga >> 32) & 0x01FFFFFFu) | 0x80000000u));
  u32x4 g0 = {g00, g01, g02, g03};
  // group1: data_size=1 (2B), tensor_dim0=row_stride, tensor_dim1=tile_h,
  //         tile_dim0=tile_w, tile_dim1=tile_h, dim0_stride=row_stride
  unsigned td0 = row_stride_elems, td1 = tile_h;
  int g10 = 0x00010000;
  int g11 = __builtin_amdgcn_readfirstlane((int)((td0 & 0xFFFFu) << 16));
  int g12 = __builtin_amdgcn_readfirstlane((int)(((td0 >> 16) & 0xFFFFu) | ((td1 & 0xFFFFu) << 16)));
  int g13 = __builtin_amdgcn_readfirstlane((int)(((td1 >> 16) & 0xFFFFu) | ((tile_w & 0xFFFFu) << 16)));
  int g14 = __builtin_amdgcn_readfirstlane((int)(td1 & 0xFFFFu));
  int g15 = __builtin_amdgcn_readfirstlane((int)row_stride_elems);
  i32x8 g1 = {g10, g11, g12, g13, g14, g15, 0, 0};
  i32x4 z4 = {0, 0, 0, 0};
  i32x8 z8 = {0, 0, 0, 0, 0, 0, 0, 0};
  __builtin_amdgcn_tensor_load_to_lds(g0, g1, z4, z4, z8, 0);
}
#endif

// ---------------------------------------------------------------------------
// K1: depthwise 3x3 conv + bias -> BN -> ReLU   (Q branch front end, VALU)
// ---------------------------------------------------------------------------
__global__ void k_qdw(const float* __restrict__ x, const float* __restrict__ w,
                      const float* __restrict__ b,
                      const float* __restrict__ g, const float* __restrict__ be,
                      const float* __restrict__ mn, const float* __restrict__ vr,
                      float* __restrict__ out) {
  int idx = blockIdx.x * blockDim.x + threadIdx.x;   // over B*C*H*W
  int xc = idx & 63;
  int y  = (idx >> 6) & 63;
  int c  = (idx >> 12) & 255;
  const float* xp = x + (size_t)(idx >> 12) * 4096;
  const float* wp = w + c * 9;
  float acc = b[c];
#pragma unroll
  for (int ky = 0; ky < 3; ++ky) {
    int yy = y + ky - 1;
    if (yy < 0 || yy > 63) continue;
#pragma unroll
    for (int kx = 0; kx < 3; ++kx) {
      int xx = xc + kx - 1;
      if (xx < 0 || xx > 63) continue;
      acc = fmaf(wp[ky * 3 + kx], xp[(yy << 6) + xx], acc);
    }
  }
  float inv = rsqrtf(vr[c] + EPSV);
  float v = (acc - mn[c]) * (g[c] * inv) + be[c];
  out[idx] = fmaxf(v, 0.0f);
}

// ---------------------------------------------------------------------------
// K2: generic small direct conv (fp32, VALU) writing into a concat buffer
// ---------------------------------------------------------------------------
__global__ void k_conv(const float* __restrict__ x, const float* __restrict__ w,
                       const float* __restrict__ b, float* __restrict__ out,
                       int Cout, int KH, int KW, int padH, int padW, int coOff) {
  int idx = blockIdx.x * blockDim.x + threadIdx.x;   // over B*Cout*4096
  int sp = idx & 4095;
  int xc = sp & 63;
  int y  = sp >> 6;
  int t  = idx >> 12;
  int co = t % Cout;
  int bb = t / Cout;
  float acc = b[co];
  const float* xb = x + (size_t)bb * CCH * 4096;
  const float* wb = w + (size_t)co * CCH * KH * KW;
  for (int ci = 0; ci < CCH; ++ci) {
    const float* xp = xb + (size_t)ci * 4096;
    const float* wq = wb + ci * KH * KW;
    __builtin_prefetch(xp + 4096, 0, 0);   // global_prefetch_b8 next plane
    for (int ky = 0; ky < KH; ++ky) {
      int yy = y + ky - padH;
      if (yy < 0 || yy > 63) continue;
      for (int kx = 0; kx < KW; ++kx) {
        int xx = xc + kx - padW;
        if (xx < 0 || xx > 63) continue;
        acc = fmaf(wq[ky * KW + kx], xp[(yy << 6) + xx], acc);
      }
    }
  }
  out[((size_t)bb * CCH + coOff + co) * 4096 + sp] = acc;
}

// ---------------------------------------------------------------------------
// K3: 1x1 conv as WMMA GEMM.  One wave per (coTile x tokTile).
//   mode 0: fp32 CHW (concat buffer, channel offset)
//   mode 1: f16  CHW (K map, attention B-matrix layout, k=ci)
//   mode 2: f16  token-major (Q / V maps, [tok][ch])
// ---------------------------------------------------------------------------
__global__ void k_gemm1x1(const float* __restrict__ in, const float* __restrict__ w,
                          const float* __restrict__ bias,
                          const float* __restrict__ bn_g, const float* __restrict__ bn_b,
                          const float* __restrict__ bn_m, const float* __restrict__ bn_v,
                          int Cout, int coOff,
                          float* __restrict__ out_f32, _Float16* __restrict__ out_f16,
                          int mode) {
  const int lane = threadIdx.x;          // 32 threads = 1 wave
  const int bb   = blockIdx.z;
  const int coT  = blockIdx.y;
  const int tokT = blockIdx.x;
  const int half = lane >> 4;
  const int nIdx = lane & 15;
  const int tok  = tokT * 16 + nIdx;
  const int mrow = coT * 16 + nIdx;               // A-matrix row (out channel)
  const int mc   = mrow < Cout ? mrow : Cout - 1; // branchless pad handling
  const float msk = mrow < Cout ? 1.0f : 0.0f;
  const float* wrow = w + (size_t)mc * CCH;
  const float* inB  = in + (size_t)bb * CCH * 4096;

  v8f acc = {};
#pragma unroll
  for (int ch = 0; ch < 8; ++ch) {       // 8 k-chunks of 32 over Cin=256
    v16h a, bm;
#pragma unroll
    for (int r = 0; r < 8; ++r) {
      int kk = ch * 32 + (r >> 2) * 16 + half * 8 + (r & 3) * 2;
      a[2 * r]     = (_Float16)(wrow[kk] * msk);
      a[2 * r + 1] = (_Float16)(wrow[kk + 1] * msk);
      bm[2 * r]     = (_Float16)inB[(size_t)kk * 4096 + tok];
      bm[2 * r + 1] = (_Float16)inB[(size_t)(kk + 1) * 4096 + tok];
    }
    acc = __builtin_amdgcn_wmma_f32_16x16x32_f16(false, a, false, bm,
                                                 (short)0, acc, false, false);
  }

#pragma unroll
  for (int r = 0; r < 8; ++r) {
    int co = coT * 16 + r + 8 * half;    // C/D layout: M = r + 8*(lane>>4)
    if (co >= Cout) continue;
    float v = acc[r] + bias[co];
    if (bn_g != nullptr) {
      float inv = rsqrtf(bn_v[co] + EPSV);
      v = (v - bn_m[co]) * (bn_g[co] * inv) + bn_b[co];
    }
    if (mode == 0)
      out_f32[((size_t)bb * CCH + coOff + co) * 4096 + tok] = v;
    else if (mode == 1)
      out_f16[((size_t)bb * CCH + co) * 4096 + tok] = (_Float16)v;
    else
      out_f16[((size_t)bb * NTOK + tok) * CCH + co] = (_Float16)v;
  }
}

// ---------------------------------------------------------------------------
// K4: spatial self-attention, 16 query rows / workgroup, 8 waves.
//   Q strip + K tiles staged into LDS by the Tensor Data Mover (TDM),
//   double-buffered per wave with s_wait_tensorcnt.
//   Phase A: S = (Q Kt) * scale -> f16 strip [16][4096] in LDS  (WMMA)
//   Phase B: softmax in-place (row max / exp / row sum via wave32 shfl)
//   Phase C: O = P V, divide by row sum, write f16 token-major   (WMMA)
// ---------------------------------------------------------------------------
__global__ void k_attn(const _Float16* __restrict__ Qt, const _Float16* __restrict__ Kc,
                       const _Float16* __restrict__ Vt, _Float16* __restrict__ Ot) {
  extern __shared__ char smem[];
  _Float16* Sh   = (_Float16*)(smem + LDS_SH);
  _Float16* Qs   = (_Float16*)(smem + LDS_QS);
  float*    rowL = (float*)(smem + LDS_ROWL);

  const int bb   = blockIdx.y;
  const int qT   = blockIdx.x;            // 16-row query tile
  const int tid  = threadIdx.x;
  const int lane = tid & 31;
  const int wave = tid >> 5;
  const int half = lane >> 4;
  const int nIdx = lane & 15;

  // ---- stage Q strip (16 tok x 256 ci, contiguous 8KB) into LDS ----
  const _Float16* Qg = Qt + ((size_t)bb * NTOK + (size_t)qT * 16) * CCH;
#ifdef HAVE_TDM
  unsigned ldsbase = (unsigned)(uintptr_t)(void*)smem;
  if (wave == 0) {
    tdm_load_2d_f16(ldsbase + LDS_QS, Qg, 4096, 1, 4096);
    __builtin_amdgcn_s_wait_tensorcnt(0);
  }
#else
  for (int t = tid; t < 2048; t += 256)
    ((unsigned*)Qs)[t] = ((const unsigned*)Qg)[t];
#endif
  __syncthreads();

  // ---- build Q A-operands from LDS (8 chunks x v16h) ----
  v16h aq[8];
#pragma unroll
  for (int ch = 0; ch < 8; ++ch) {
#pragma unroll
    for (int r = 0; r < 8; ++r) {
      int kk = ch * 32 + (r >> 2) * 16 + half * 8 + (r & 3) * 2;
      aq[ch][2 * r]     = Qs[nIdx * CCH + kk];
      aq[ch][2 * r + 1] = Qs[nIdx * CCH + kk + 1];
    }
  }

  // ---- Phase A: S tiles, waves stride over 256 key tiles ----
  const _Float16* Kb = Kc + (size_t)bb * CCH * NTOK;
#ifdef HAVE_TDM
  {
    unsigned myOff = LDS_KST + (unsigned)wave * 16384u;   // 2 x 8KB buffers
    tdm_load_2d_f16(ldsbase + myOff, Kb + wave * 16, 16, 256, NTOK);
    int i = 0;
    for (int kt = wave; kt < 256; kt += 8, ++i) {
      int buf = i & 1;
      if (kt + 8 < 256) {
        tdm_load_2d_f16(ldsbase + myOff + (unsigned)(buf ^ 1) * 8192u,
                        Kb + (kt + 8) * 16, 16, 256, NTOK);
        __builtin_amdgcn_s_wait_tensorcnt(1);   // current buffer complete
      } else {
        __builtin_amdgcn_s_wait_tensorcnt(0);
      }
      const _Float16* st = (const _Float16*)(smem + myOff + (unsigned)buf * 8192u);
      v8f c = {};
#pragma unroll
      for (int ch = 0; ch < 8; ++ch) {
        v16h bm;
#pragma unroll
        for (int r = 0; r < 8; ++r) {
          int kk = ch * 32 + (r >> 2) * 16 + half * 8 + (r & 3) * 2;
          bm[2 * r]     = st[kk * 16 + nIdx];     // staged [ci][16 keys]
          bm[2 * r + 1] = st[(kk + 1) * 16 + nIdx];
        }
        c = __builtin_amdgcn_wmma_f32_16x16x32_f16(false, aq[ch], false, bm,
                                                   (short)0, c, false, false);
      }
#pragma unroll
      for (int r = 0; r < 8; ++r) {
        int row = r + 8 * half;
        Sh[row * NTOK + kt * 16 + nIdx] = (_Float16)(c[r] * ATTN_SCALE);
      }
    }
  }
#else
  for (int kt = wave; kt < 256; kt += 8) {
    int key = kt * 16 + nIdx;
    v8f c = {};
#pragma unroll
    for (int ch = 0; ch < 8; ++ch) {
      v16h bm;
#pragma unroll
      for (int r = 0; r < 8; ++r) {
        int kk = ch * 32 + (r >> 2) * 16 + half * 8 + (r & 3) * 2;
        bm[2 * r]     = Kb[(size_t)kk * NTOK + key];
        bm[2 * r + 1] = Kb[(size_t)(kk + 1) * NTOK + key];
      }
      c = __builtin_amdgcn_wmma_f32_16x16x32_f16(false, aq[ch], false, bm,
                                                 (short)0, c, false, false);
    }
#pragma unroll
    for (int r = 0; r < 8; ++r) {
      int row = r + 8 * half;
      Sh[row * NTOK + kt * 16 + nIdx] = (_Float16)(c[r] * ATTN_SCALE);
    }
  }
#endif
  __syncthreads();

  // ---- Phase B: softmax in place (16 rows x 16 threads each) ----
  {
    int row = tid >> 4;
    int j   = tid & 15;
    float m = -3.0e38f;
    for (int k = j; k < NTOK; k += 16)
      m = fmaxf(m, (float)Sh[row * NTOK + k]);
    m = fmaxf(m, __shfl_xor(m, 1));
    m = fmaxf(m, __shfl_xor(m, 2));
    m = fmaxf(m, __shfl_xor(m, 4));
    m = fmaxf(m, __shfl_xor(m, 8));
    float s = 0.0f;
    for (int k = j; k < NTOK; k += 16) {
      float e = __expf((float)Sh[row * NTOK + k] - m);
      Sh[row * NTOK + k] = (_Float16)e;
      s += e;
    }
    s += __shfl_xor(s, 1);
    s += __shfl_xor(s, 2);
    s += __shfl_xor(s, 4);
    s += __shfl_xor(s, 8);
    if (j == 0) rowL[row] = s;
  }
  __syncthreads();

  // ---- Phase C: O = P * V, each wave owns 2 channel tiles ----
  const _Float16* Vb = Vt + (size_t)bb * NTOK * CCH;
  for (int cti = 0; cti < 2; ++cti) {
    int ct = wave + cti * 8;
    int chcol = ct * 16 + nIdx;
    v8f c = {};
    for (int ch = 0; ch < 128; ++ch) {   // 128 key chunks of 32
      v16h a, bm;
#pragma unroll
      for (int r = 0; r < 8; ++r) {
        int kk = ch * 32 + (r >> 2) * 16 + half * 8 + (r & 3) * 2;
        a[2 * r]     = Sh[nIdx * NTOK + kk];        // P in A-layout (M = query row)
        a[2 * r + 1] = Sh[nIdx * NTOK + kk + 1];
        bm[2 * r]     = Vb[(size_t)kk * CCH + chcol];
        bm[2 * r + 1] = Vb[(size_t)(kk + 1) * CCH + chcol];
      }
      c = __builtin_amdgcn_wmma_f32_16x16x32_f16(false, a, false, bm,
                                                 (short)0, c, false, false);
    }
#pragma unroll
    for (int r = 0; r < 8; ++r) {
      int row = r + 8 * half;
      float invl = 1.0f / rowL[row];
      Ot[((size_t)bb * NTOK + (size_t)qT * 16 + row) * CCH + chcol] = (_Float16)(c[r] * invl);
    }
  }
}

// ---------------------------------------------------------------------------
// K5: out projection (1x1 conv over attention output) + bias + residual
// ---------------------------------------------------------------------------
__global__ void k_outproj(const _Float16* __restrict__ Ot, const float* __restrict__ w,
                          const float* __restrict__ bias, const float* __restrict__ x,
                          float* __restrict__ out) {
  const int lane = threadIdx.x;          // 1 wave
  const int bb   = blockIdx.z;
  const int coT  = blockIdx.y;
  const int tokT = blockIdx.x;
  const int half = lane >> 4;
  const int nIdx = lane & 15;
  const int co   = coT * 16 + nIdx;      // B-matrix column
  const _Float16* Ob = Ot + ((size_t)bb * NTOK + (size_t)tokT * 16 + nIdx) * CCH;

  v8f c = {};
#pragma unroll
  for (int ch = 0; ch < 8; ++ch) {
    v16h a, bm;
#pragma unroll
    for (int r = 0; r < 8; ++r) {
      int kk = ch * 32 + (r >> 2) * 16 + half * 8 + (r & 3) * 2;
      a[2 * r]     = Ob[kk];
      a[2 * r + 1] = Ob[kk + 1];
      bm[2 * r]     = (_Float16)w[(size_t)co * CCH + kk];
      bm[2 * r + 1] = (_Float16)w[(size_t)co * CCH + kk + 1];
    }
    c = __builtin_amdgcn_wmma_f32_16x16x32_f16(false, a, false, bm,
                                               (short)0, c, false, false);
  }
#pragma unroll
  for (int r = 0; r < 8; ++r) {
    int tok = tokT * 16 + r + 8 * half;  // C/D layout: M = token
    int coc = coT * 16 + nIdx;
    size_t idx = ((size_t)bb * CCH + coc) * 4096 + tok;
    out[idx] = c[r] + bias[coc] + x[idx];
  }
}

// ---------------------------------------------------------------------------
// Host-side launch
// ---------------------------------------------------------------------------
extern "C" void kernel_launch(void* const* d_in, const int* in_sizes, int n_in,
                              void* d_out, int out_size, void* d_ws, size_t ws_size,
                              hipStream_t stream) {
  const float* x      = (const float*)d_in[0];
  const float* qdw_w  = (const float*)d_in[1];
  const float* qdw_b  = (const float*)d_in[2];
  const float* qbn_g  = (const float*)d_in[3];
  const float* qbn_b  = (const float*)d_in[4];
  const float* qbn_m  = (const float*)d_in[5];
  const float* qbn_v  = (const float*)d_in[6];
  const float* qpw_w  = (const float*)d_in[7];
  const float* qpw_b  = (const float*)d_in[8];
  const float* kc3_w  = (const float*)d_in[9];
  const float* kc3_b  = (const float*)d_in[10];
  const float* kc5_w  = (const float*)d_in[11];
  const float* kc5_b  = (const float*)d_in[12];
  const float* kc7_w  = (const float*)d_in[13];
  const float* kc7_b  = (const float*)d_in[14];
  const float* kf_w   = (const float*)d_in[15];
  const float* kf_b   = (const float*)d_in[16];
  const float* vh_w   = (const float*)d_in[17];
  const float* vh_b   = (const float*)d_in[18];
  const float* vv_w   = (const float*)d_in[19];
  const float* vv_b   = (const float*)d_in[20];
  const float* vd1_w  = (const float*)d_in[21];
  const float* vd1_b  = (const float*)d_in[22];
  const float* vd2_w  = (const float*)d_in[23];
  const float* vd2_b  = (const float*)d_in[24];
  const float* vf_w   = (const float*)d_in[25];
  const float* vf_b   = (const float*)d_in[26];
  const float* vbn_g  = (const float*)d_in[27];
  const float* vbn_b  = (const float*)d_in[28];
  const float* vbn_m  = (const float*)d_in[29];
  const float* vbn_v  = (const float*)d_in[30];
  const float* out_w  = (const float*)d_in[31];
  const float* out_b  = (const float*)d_in[32];

  char* ws = (char*)d_ws;
  const size_t MAP32 = (size_t)BSZ * CCH * NTOK * 4;   // 8 MB
  const size_t MAP16 = (size_t)BSZ * CCH * NTOK * 2;   // 4 MB
  float*     qmid = (float*)ws;
  float*     kcat = (float*)(ws + MAP32);
  float*     vcat = (float*)(ws + 2 * MAP32);
  _Float16*  Qt   = (_Float16*)(ws + 3 * MAP32);
  _Float16*  Kc   = (_Float16*)(ws + 3 * MAP32 + MAP16);
  _Float16*  Vt   = (_Float16*)(ws + 3 * MAP32 + 2 * MAP16);
  _Float16*  Ot   = (_Float16*)(ws + 3 * MAP32 + 3 * MAP16);   // total 40 MB

  // Q branch front end
  k_qdw<<<(BSZ * CCH * NTOK) / 256, 256, 0, stream>>>(
      x, qdw_w, qdw_b, qbn_g, qbn_b, qbn_m, qbn_v, qmid);

  // K branch: 1x1 (85) via WMMA GEMM, 3x3 (85) and 5x5 (86) direct
  k_gemm1x1<<<dim3(256, 6, BSZ), 32, 0, stream>>>(
      x, kc3_w, kc3_b, nullptr, nullptr, nullptr, nullptr, 85, 0, kcat, nullptr, 0);
  k_conv<<<(BSZ * 85 * NTOK) / 256, 256, 0, stream>>>(x, kc5_w, kc5_b, kcat, 85, 3, 3, 1, 1, 85);
  k_conv<<<(BSZ * 86 * NTOK) / 256, 256, 0, stream>>>(x, kc7_w, kc7_b, kcat, 86, 5, 5, 2, 2, 170);

  // V branch: four directional convs into concat buffer
  k_conv<<<(BSZ * 64 * NTOK) / 256, 256, 0, stream>>>(x, vh_w,  vh_b,  vcat, 64, 1, 3, 0, 1, 0);
  k_conv<<<(BSZ * 64 * NTOK) / 256, 256, 0, stream>>>(x, vv_w,  vv_b,  vcat, 64, 3, 1, 1, 0, 64);
  k_conv<<<(BSZ * 64 * NTOK) / 256, 256, 0, stream>>>(x, vd1_w, vd1_b, vcat, 64, 3, 3, 1, 1, 128);
  k_conv<<<(BSZ * 64 * NTOK) / 256, 256, 0, stream>>>(x, vd2_w, vd2_b, vcat, 64, 3, 3, 1, 1, 192);

  // 1x1 fuse GEMMs -> attention-ready f16 layouts
  dim3 gfull(256, 16, BSZ);
  k_gemm1x1<<<gfull, 32, 0, stream>>>(qmid, qpw_w, qpw_b,
      nullptr, nullptr, nullptr, nullptr, 256, 0, nullptr, Qt, 2);   // token-major
  k_gemm1x1<<<gfull, 32, 0, stream>>>(kcat, kf_w, kf_b,
      nullptr, nullptr, nullptr, nullptr, 256, 0, nullptr, Kc, 1);   // channel-major
  k_gemm1x1<<<gfull, 32, 0, stream>>>(vcat, vf_w, vf_b,
      vbn_g, vbn_b, vbn_m, vbn_v, 256, 0, nullptr, Vt, 2);           // token-major + BN

  // attention: 16 query rows per WG; S strip + Q stage + TDM K buffers in LDS
  k_attn<<<dim3(NTOK / 16, BSZ), 256, LDS_TOTAL, stream>>>(Qt, Kc, Vt, Ot);

  // out projection + residual -> d_out (fp32 CHW)
  k_outproj<<<dim3(256, 16, BSZ), 32, 0, stream>>>(Ot, out_w, out_b, x, (float*)d_out);
}
